// LeakyRNNModel_10033043604057
// MI455X (gfx1250) — matrix operand order repaired and checked by reading, not compiled, not run on hardware
//
#include <hip/hip_runtime.h>
#include <hip/hip_bf16.h>

// ---------------- types ----------------
typedef __bf16 v16bf __attribute__((ext_vector_type(16)));
typedef float  v8f   __attribute__((ext_vector_type(8)));
typedef unsigned int u32x4 __attribute__((ext_vector_type(4)));
typedef int    i32x8 __attribute__((ext_vector_type(8)));
typedef int    i32x4 __attribute__((ext_vector_type(4)));

// ---------------- problem constants ----------------
#define N_HID   1024
#define NIN     32
#define NOUT    32
#define BATCH   16
#define TSTEPS  1000
#define ALPHA_C 0.1f
#define SNOISE  0.01f
#define NWG     16          // persistent workgroups, 64 hidden cols each

#define SLOT_ELEMS (BATCH * N_HID)            // 16384 bf16 per r-slot (one WMMA M-tile, packed A layout)
// workspace byte offsets
#define WEFF_OFF 0
#define WEFF_BYTES ((size_t)N_HID * N_HID * 2)                 // 2 MB packed bf16
#define WOUT_OFF (WEFF_OFF + WEFF_BYTES)
#define WOUT_BYTES ((size_t)N_HID * NOUT * 2)                  // 64 KB packed bf16
#define RSEQ_OFF (WOUT_OFF + WOUT_BYTES)
#define RSEQ_BYTES ((size_t)(TSTEPS + 1) * SLOT_ELEMS * 2)     // ~31.3 MB
#define CNT_OFF  (RSEQ_OFF + RSEQ_BYTES)

// LDS layout inside the persistent kernel (dynamic shared)
#define LDS_WEFF_BYTES (64 * 1024 * 2)        // 128 KB W_eff slice
#define LDS_WIN_BYTES  (NIN * 64 * 4)         // 8 KB W_in slice
#define LDS_BIAS_BYTES (64 * 4)               // bias + b_out@W_fb folded
#define LDS_INP_BYTES  (BATCH * NIN * 4)      // 2 KB staged inputs for step t
#define LDS_TOTAL (LDS_WEFF_BYTES + LDS_WIN_BYTES + LDS_BIAS_BYTES + LDS_INP_BYTES)

// ---------------------------------------------------------------------------
// init: zero r-slot 0 (r_0 = tanh(0) = 0) and the barrier counter
// ---------------------------------------------------------------------------
__global__ void init_ws(__bf16* __restrict__ rseq, int* __restrict__ cnt) {
    int id = blockIdx.x * blockDim.x + threadIdx.x;
    if (id < SLOT_ELEMS) rseq[id] = (__bf16)0.0f;
    if (id == 0) *cnt = 0;
}

// ---------------------------------------------------------------------------
// pack W_eff = W_rec + W_out @ W_fb  into WMMA B-fragment layout (bf16)
// layout: [ct(64)][kb(32)][lane(32)][idx(16)]; lane = n%16 + 16*k[4], idx = k&15
// ---------------------------------------------------------------------------
__global__ void pack_weff(const float* __restrict__ W_rec,
                          const float* __restrict__ W_out,
                          const float* __restrict__ W_fb,
                          __bf16* __restrict__ weff_pk) {
    int id = blockIdx.x * blockDim.x + threadIdx.x;    // 1M elements
    int k  = id >> 10;            // source hidden index (K)
    int nt = id & 1023;           // dest hidden index (N)
    float acc = W_rec[k * N_HID + nt];
    #pragma unroll
    for (int o = 0; o < NOUT; ++o)
        acc += W_out[k * NOUT + o] * W_fb[o * N_HID + nt];
    int ct = nt >> 4, nl = nt & 15;
    int kb = k >> 5,  kk = k & 31;
    int lanep = nl + (((kk >> 4) & 1) << 4);
    int idxp  = kk & 15;
    weff_pk[((size_t)(ct * 32 + kb) * 32 + lanep) * 16 + idxp] = (__bf16)acc;
}

// ---------------------------------------------------------------------------
// pack W_out [1024][32] into WMMA B-fragment layout (2 column tiles)
// ---------------------------------------------------------------------------
__global__ void pack_wout(const float* __restrict__ W_out, __bf16* __restrict__ wout_pk) {
    int id = blockIdx.x * blockDim.x + threadIdx.x;    // 32768 elements
    int k = id >> 5, o = id & 31;
    int ct = o >> 4, nl = o & 15;
    int kb = k >> 5, kk = k & 31;
    int lanep = nl + (((kk >> 4) & 1) << 4);
    int idxp  = kk & 15;
    wout_pk[(size_t)ct * (32 * 32 * 16) + ((kb * 32 + lanep) * 16) + idxp] = (__bf16)W_out[k * NOUT + o];
}

// ---------------------------------------------------------------------------
// persistent recurrent kernel: 16 WGs x 128 threads (4 waves), wave w owns one
// 16-column tile. W_eff slice lives in LDS (TDM-loaded once). x state lives in
// registers in WMMA C layout. Cross-WG sync via agent-scope atomic counter.
// ---------------------------------------------------------------------------
__global__ void __launch_bounds__(128) rnn_persistent(
        const float* __restrict__ inputs, const float* __restrict__ noise,
        const float* __restrict__ W_in,   const float* __restrict__ bias,
        const float* __restrict__ b_out,  const float* __restrict__ W_fb,
        const __bf16* __restrict__ weff_pk, __bf16* __restrict__ rseq,
        int* __restrict__ cnt) {
    extern __shared__ char lds[];
    __bf16* weff_lds  = (__bf16*)lds;                                  // 128 KB
    float*  win_lds   = (float*)(lds + LDS_WEFF_BYTES);                // [32][64]
    float*  biasc_lds = (float*)(lds + LDS_WEFF_BYTES + LDS_WIN_BYTES);// [64]
    float*  inp_lds   = (float*)(lds + LDS_WEFF_BYTES + LDS_WIN_BYTES + LDS_BIAS_BYTES); // [16][32]

    const int wg   = blockIdx.x;
    const int tid  = threadIdx.x;
    const int lane = tid & 31;
    const int wv   = tid >> 5;             // 0..3 -> column tile within this WG

    // ---- TDM: pull this WG's contiguous 128KB packed W_eff slice into LDS ----
    if (wv == 0) {
        unsigned long long ga = (unsigned long long)(size_t)weff_pk
                              + (unsigned long long)wg * (unsigned long long)LDS_WEFF_BYTES;
        u32x4 g0;
        g0[0] = 1u;                                            // count=1, user descriptor
        g0[1] = 0u;                                            // lds_addr = 0 (start of dynamic LDS)
        g0[2] = (unsigned)(ga & 0xFFFFFFFFull);                // global_addr[31:0]
        g0[3] = (unsigned)((ga >> 32) & 0x01FFFFFFull) | (2u << 30); // addr[56:32] | type=2
        i32x8 g1;
        g1[0] = (3 << 16);                  // workgroup_mask=0, data_size=8B
        g1[1] = (int)(16384u << 16);        // tensor_dim0[15:0] = 16384 (8B elems)
        g1[2] = (int)(1u << 16);            // tensor_dim0[31:16]=0, tensor_dim1=1
        g1[3] = (int)(16384u << 16);        // tile_dim0 = 16384
        g1[4] = 1;                          // tile_dim1 = 1
        g1[5] = 16384;                      // tensor_dim0_stride
        g1[6] = 0; g1[7] = 0;
        i32x4 gz4 = {0, 0, 0, 0};
        i32x8 gz8 = {0, 0, 0, 0, 0, 0, 0, 0};
        __builtin_amdgcn_tensor_load_to_lds(g0, g1, gz4, gz4, gz8, 0);
        __builtin_amdgcn_s_wait_tensorcnt(0);
    }

    // ---- cooperative load of W_in slice and folded bias (bias + b_out@W_fb) ----
    for (int e = tid; e < NIN * 64; e += 128) {
        int j = e >> 6, c = e & 63;
        win_lds[j * 64 + c] = W_in[j * N_HID + wg * 64 + c];
    }
    if (tid < 64) {
        int n = wg * 64 + tid;
        float a = bias[n];
        #pragma unroll
        for (int o = 0; o < NOUT; ++o) a += b_out[o] * W_fb[o * N_HID + n];
        biasc_lds[tid] = a;
    }
    __syncthreads();

    // ---- per-lane invariants ----
    const int ct    = wv;
    const int nloc  = ct * 16 + (lane & 15);          // local column 0..63
    const int nglob = wg * 64 + nloc;                 // global hidden index
    const int mhi   = (lane >> 4) << 3;               // +8 batch rows for upper half-wave
    // packed-A store coordinates for column nglob (r produced here)
    const int kk_w     = nglob & 31;
    const int kb_w     = nglob >> 5;
    const int st_idx   = kk_w & 15;
    const int st_lane0 = ((kk_w >> 4) & 1) << 4;

    float wreg[NIN];                                   // W_in column for this lane
    #pragma unroll
    for (int j = 0; j < NIN; ++j) wreg[j] = win_lds[j * 64 + nloc];
    const float ubias = biasc_lds[nloc];

    float xr[8];
    #pragma unroll
    for (int i = 0; i < 8; ++i) xr[i] = 0.0f;

    const __bf16* bfrag = weff_lds + (size_t)ct * (32 * 32 * 16);

    for (int t = 0; t < TSTEPS; ++t) {
        // stage this step's input rows (16 x 32 f32) into LDS once
        for (int e = tid; e < BATCH * NIN; e += 128) {
            int bb = e >> 5, j = e & 31;
            inp_lds[e] = inputs[((size_t)bb * TSTEPS + t) * NIN + j];
        }
        __syncthreads();

        // u_rec tile = r_t[16x1024] @ W_eff[:, 16 cols]   (32 chained WMMAs)
        const __bf16* aslot = rseq + (size_t)t * SLOT_ELEMS;
        v8f c = {};
        #pragma unroll 4
        for (int kb = 0; kb < 32; ++kb) {
            v16bf a = *(const v16bf*)(aslot + (size_t)kb * 512 + lane * 16);
            v16bf b = *(const v16bf*)(bfrag + ((size_t)kb * 32 + lane) * 16);
            c = __builtin_amdgcn_wmma_f32_16x16x32_bf16(false, a, false, b,
                                                        (short)0, c, false, false);
        }

        // elementwise leaky update + tanh, write r_{t+1} in packed A layout
        __bf16* wslot = rseq + (size_t)(t + 1) * SLOT_ELEMS;
        #pragma unroll
        for (int i = 0; i < 8; ++i) {
            int bb = i + mhi;                                  // batch row
            float u = ubias + c[i];
            #pragma unroll
            for (int j = 0; j < NIN; ++j) u += inp_lds[bb * NIN + j] * wreg[j];
            float nz = noise[((size_t)bb * TSTEPS + t) * N_HID + nglob];
            float xn = (1.0f - ALPHA_C) * xr[i] + ALPHA_C * u + SNOISE * nz;
            xr[i] = xn;
            float r = tanhf(xn);
            wslot[(size_t)kb_w * 512 + (bb + st_lane0) * 16 + st_idx] = (__bf16)r;
        }

        // ---- inter-workgroup barrier (release stores, acquire next r) ----
        __syncthreads();
        if (tid == 0) {
            __hip_atomic_fetch_add(cnt, 1, __ATOMIC_RELEASE, __HIP_MEMORY_SCOPE_AGENT);
            const int target = NWG * (t + 1);
            while (__hip_atomic_load(cnt, __ATOMIC_ACQUIRE, __HIP_MEMORY_SCOPE_AGENT) < target) {
                __builtin_amdgcn_s_sleep(1);
            }
        }
        __builtin_amdgcn_s_cluster_barrier();   // NOP unless cluster-dispatched
        __syncthreads();
    }
}

// ---------------------------------------------------------------------------
// readout: out[b][t][:] = r_{t+1} @ W_out + b_out. One wave per (t, col-tile).
// ---------------------------------------------------------------------------
__global__ void __launch_bounds__(32) readout(
        const __bf16* __restrict__ rseq, const __bf16* __restrict__ wout_pk,
        const float* __restrict__ b_out, float* __restrict__ out) {
    const int t    = blockIdx.x >> 1;
    const int ct   = blockIdx.x & 1;
    const int lane = threadIdx.x & 31;
    const __bf16* aslot = rseq + (size_t)(t + 1) * SLOT_ELEMS;
    const __bf16* bbase = wout_pk + (size_t)ct * (32 * 32 * 16);
    v8f c = {};
    #pragma unroll 4
    for (int kb = 0; kb < 32; ++kb) {
        v16bf a = *(const v16bf*)(aslot + (size_t)kb * 512 + lane * 16);
        v16bf b = *(const v16bf*)(bbase + ((size_t)kb * 32 + lane) * 16);
        c = __builtin_amdgcn_wmma_f32_16x16x32_bf16(false, a, false, b,
                                                    (short)0, c, false, false);
    }
    const int o   = ct * 16 + (lane & 15);
    const float bo = b_out[o];
    const int mhi = (lane >> 4) << 3;
    #pragma unroll
    for (int i = 0; i < 8; ++i) {
        int bb = i + mhi;
        out[((size_t)bb * TSTEPS + t) * NOUT + o] = c[i] + bo;
    }
}

// ---------------------------------------------------------------------------
extern "C" void kernel_launch(void* const* d_in, const int* in_sizes, int n_in,
                              void* d_out, int out_size, void* d_ws, size_t ws_size,
                              hipStream_t stream) {
    (void)in_sizes; (void)n_in; (void)out_size; (void)ws_size;
    const float* inputs = (const float*)d_in[0];
    const float* noise  = (const float*)d_in[1];
    const float* W_in   = (const float*)d_in[2];
    const float* bias   = (const float*)d_in[3];
    const float* W_rec  = (const float*)d_in[4];
    const float* W_out  = (const float*)d_in[5];
    const float* b_out  = (const float*)d_in[6];
    const float* W_fb   = (const float*)d_in[7];
    float* out = (float*)d_out;

    char* ws = (char*)d_ws;
    __bf16* weff_pk = (__bf16*)(ws + WEFF_OFF);
    __bf16* wout_pk = (__bf16*)(ws + WOUT_OFF);
    __bf16* rseq    = (__bf16*)(ws + RSEQ_OFF);
    int*    cnt     = (int*)(ws + CNT_OFF);

    (void)hipFuncSetAttribute(reinterpret_cast<const void*>(rnn_persistent),
                              hipFuncAttributeMaxDynamicSharedMemorySize, (int)LDS_TOTAL);

    init_ws<<<(SLOT_ELEMS + 255) / 256, 256, 0, stream>>>(rseq, cnt);
    pack_weff<<<(N_HID * N_HID) / 256, 256, 0, stream>>>(W_rec, W_out, W_fb, weff_pk);
    pack_wout<<<(N_HID * NOUT) / 256, 256, 0, stream>>>(W_out, wout_pk);
    rnn_persistent<<<NWG, 128, LDS_TOTAL, stream>>>(inputs, noise, W_in, bias,
                                                    b_out, W_fb, weff_pk, rseq, cnt);
    readout<<<2 * TSTEPS, 32, 0, stream>>>(rseq, wout_pk, b_out, out);
}
